// MemoryRead_21320217657495
// MI455X (gfx1250) — matrix operand order repaired and checked by reading, not compiled, not run on hardware
//
#include <hip/hip_runtime.h>
#include <hip/hip_bf16.h>
#include <math.h>

#define N_  2048
#define M_  256
#define DQ_ 512
#define DM_ 512
#define P_  256

typedef __attribute__((ext_vector_type(16))) __bf16 v16bf;
typedef __attribute__((ext_vector_type(8)))  float  v8f;

union BFrag { uint4 u[2]; v16bf v; };

static __device__ __forceinline__ unsigned short bfbits(float f) {
  union { __bf16 b; unsigned short s; } u; u.b = (__bf16)f; return u.s;
}
static __device__ __forceinline__ __bf16 tobf(float f) { return (__bf16)f; }

// Byte offset (within a 1KB fragment) of 16B chunk c for `lane`, with an
// XOR-16 swizzle so LDS ds_load_b128 across a wave is 2-way (minimal) banked.
static __device__ __forceinline__ int frag_chunk_off(int lane, int c) {
  int sw = ((lane >> 3) & 1) << 4;
  return ((lane * 32 + c * 16) ^ sw);
}

// ---------------------------------------------------------------------------
// Pack row-major [K][P] f32 matrix into WMMA 16x16x32 bf16 B fragments.
// Fragment (kt,pt) = 1KB: lane l holds col n = pt*16 + l%16 ; h = l/16 ;
// dword v holds K = kt*32 + 8h + 8*(v>>2) + {2v, 2v+1} (low 16 bits = even K).
// 16B chunks stored at frag_chunk_off() (swizzled).
// ---------------------------------------------------------------------------
__global__ void pack_b_kernel(const float* __restrict__ src,
                              unsigned* __restrict__ dst, int K, int P) {
  int idx = blockIdx.x * blockDim.x + threadIdx.x;
  int total = (K >> 5) * (P >> 4) * 256;
  if (idx >= total) return;
  int v    = idx & 7;
  int lane = (idx >> 3) & 31;
  int frag = idx >> 8;
  int PT = P >> 4;
  int pt = frag % PT;
  int kt = frag / PT;
  int p = pt * 16 + (lane & 15);
  int k = kt * 32 + ((lane >> 4) << 3) + 2 * v + ((v >> 2) << 3);
  unsigned lo = bfbits(src[(size_t)k * P + p]);
  unsigned hi = bfbits(src[(size_t)(k + 1) * P + p]);
  int boff = frag_chunk_off(lane, v >> 2) + (v & 3) * 4;  // byte within frag
  dst[frag * 256 + (boff >> 2)] = lo | (hi << 16);
}

// query f32 [N][DQ] -> bf16 into concat buffer columns [DM .. DM+DQ)
__global__ void cvt_query_kernel(const float* __restrict__ q,
                                 unsigned* __restrict__ concat_u) {
  int idx = blockIdx.x * blockDim.x + threadIdx.x;   // bf16-pair index
  if (idx >= N_ * (DQ_ / 2)) return;
  int n = idx / (DQ_ / 2);
  int d = (idx % (DQ_ / 2)) * 2;
  const float* s = q + (size_t)n * DQ_ + d;
  unsigned lo = bfbits(s[0]);
  unsigned hi = bfbits(s[1]);
  concat_u[((size_t)n * (DM_ + DQ_) + DM_ + d) >> 1] = lo | (hi << 16);
}

// ---------------------------------------------------------------------------
// qws[n][p] = (query @ W_att)[n][p] + W_att_b[p] + U_att_b[p]
// ---------------------------------------------------------------------------
__global__ void q_gemm_kernel(const __hip_bfloat16* __restrict__ concat_bf,
                              const char* __restrict__ Bpack,
                              const float* __restrict__ Wb,
                              const float* __restrict__ Ub,
                              float* __restrict__ qws) {
  int gw   = (blockIdx.x * blockDim.x + threadIdx.x) >> 5;
  int lane = threadIdx.x & 31;
  int lm = lane & 15, h = lane >> 4;
  int nt = gw >> 4;          // 0..127
  int pt = gw & 15;          // 0..15
  int co0 = frag_chunk_off(lane, 0), co1 = frag_chunk_off(lane, 1);
  const char* Arow =
      (const char*)(concat_bf + (size_t)(nt * 16 + lm) * (DM_ + DQ_) + DM_);
  v8f acc = {};
  #pragma unroll
  for (int kt = 0; kt < DQ_ / 32; ++kt) {
    BFrag a;
    a.u[0] = *(const uint4*)(Arow + (kt * 32 + 8 * h) * 2);
    a.u[1] = *(const uint4*)(Arow + (kt * 32 + 16 + 8 * h) * 2);
    BFrag b;
    const char* fb = Bpack + (size_t)(kt * (P_ / 16) + pt) * 1024;
    b.u[0] = *(const uint4*)(fb + co0);
    b.u[1] = *(const uint4*)(fb + co1);
    acc = __builtin_amdgcn_wmma_f32_16x16x32_bf16(false, a.v, false, b.v,
                                                  (short)0, acc, false, false);
  }
  int p = pt * 16 + lm;
  float bias = Wb[p] + Ub[p];
  #pragma unroll
  for (int i = 0; i < 8; ++i) {
    int row = nt * 16 + i + 8 * h;   // C layout: VGPR i, half h -> M = i + 8h
    qws[(size_t)row * P_ + p] = acc[i] + bias;
  }
}

// ---------------------------------------------------------------------------
// Fused kernel, 2 query rows per workgroup, 16 waves.
// Wave w owns memory-row tile mt=w for both n. U_att B fragments are staged
// into LDS with global_load_async_to_lds_b128 in two 128KB phases (pt 0..7,
// then 8..15); each B fragment read from LDS feeds TWO WMMAs (n0,n1).
// memory_h streamed from HBM once; subsequent sweeps hit L2.
// ---------------------------------------------------------------------------
__global__ void __launch_bounds__(512) fused_kernel(
    const float* __restrict__ memory,          // [N][M][DM] f32
    const float* __restrict__ qws,             // [N][P]  (q + both biases)
    const char* __restrict__ Upack,            // U_att B fragments (256KB)
    const float* __restrict__ w_att,           // [P]
    unsigned short* __restrict__ concat_u16) { // [N][DM+DQ] bf16 (ctx half)
  extern __shared__ char s_dyn[];              // 128KB: B frags of one phase
  __shared__ float s_q[2][P_];
  __shared__ float s_scores[2][M_];
  __shared__ float s_att[2][M_];

  int n0   = blockIdx.x * 2;
  int tid  = threadIdx.x;
  int wave = tid >> 5;       // 0..15 = m-tile
  int lane = tid & 31;
  int lm = lane & 15, h = lane >> 4;
  int co0 = frag_chunk_off(lane, 0), co1 = frag_chunk_off(lane, 1);

  if (tid < 2 * P_) s_q[tid >> 8][tid & 255] = qws[(size_t)(n0 + (tid >> 8)) * P_ + (tid & 255)];

  const float* Arow0 = memory + ((size_t)n0 * M_ + wave * 16 + lm) * DM_;
  const float* Arow1 = Arow0 + (size_t)M_ * DM_;
  unsigned sB = (unsigned)(size_t)s_dyn;

  float sacc0[8], sacc1[8];
  #pragma unroll
  for (int i = 0; i < 8; ++i) { sacc0[i] = 0.f; sacc1[i] = 0.f; }

  for (int ph = 0; ph < 2; ++ph) {
    // ---- stage 128 local fragments (kt 0..15 x ptl 0..7) into LDS (async) --
    __syncthreads();                       // prior phase consumers done
    for (int i = tid; i < 8192; i += 512) {  // 8192 x 16B = 128KB
      int l  = i >> 6;                     // local frag 0..127
      int ci = i & 63;                     // 16B chunk within frag
      int kt = l >> 3, ptl = l & 7;
      int gfrag = kt * 16 + ph * 8 + ptl;
      unsigned long long ga =
          (unsigned long long)(Upack + (size_t)gfrag * 1024 + ci * 16);
      unsigned la = sB + (unsigned)(l * 1024 + ci * 16);
      asm volatile("global_load_async_to_lds_b128 %0, %1, off"
                   :: "v"(la), "v"(ga) : "memory");
    }
    asm volatile("s_wait_asynccnt 0x0" ::: "memory");
    __syncthreads();

    // ---- GEMM: 2n x 8 P-tiles of accumulators, K sweep ---------------------
    v8f zero = {};
    v8f acc0[8], acc1[8];
    #pragma unroll
    for (int p8 = 0; p8 < 8; ++p8) { acc0[p8] = zero; acc1[p8] = zero; }

    for (int kt = 0; kt < DM_ / 32; ++kt) {
      float4 f0 = *(const float4*)(Arow0 + kt * 32 + 8 * h);
      float4 f1 = *(const float4*)(Arow0 + kt * 32 + 8 * h + 4);
      float4 f2 = *(const float4*)(Arow0 + kt * 32 + 16 + 8 * h);
      float4 f3 = *(const float4*)(Arow0 + kt * 32 + 16 + 8 * h + 4);
      v16bf a0;
      a0[0]=tobf(f0.x);  a0[1]=tobf(f0.y);  a0[2]=tobf(f0.z);  a0[3]=tobf(f0.w);
      a0[4]=tobf(f1.x);  a0[5]=tobf(f1.y);  a0[6]=tobf(f1.z);  a0[7]=tobf(f1.w);
      a0[8]=tobf(f2.x);  a0[9]=tobf(f2.y);  a0[10]=tobf(f2.z); a0[11]=tobf(f2.w);
      a0[12]=tobf(f3.x); a0[13]=tobf(f3.y); a0[14]=tobf(f3.z); a0[15]=tobf(f3.w);
      f0 = *(const float4*)(Arow1 + kt * 32 + 8 * h);
      f1 = *(const float4*)(Arow1 + kt * 32 + 8 * h + 4);
      f2 = *(const float4*)(Arow1 + kt * 32 + 16 + 8 * h);
      f3 = *(const float4*)(Arow1 + kt * 32 + 16 + 8 * h + 4);
      v16bf a1;
      a1[0]=tobf(f0.x);  a1[1]=tobf(f0.y);  a1[2]=tobf(f0.z);  a1[3]=tobf(f0.w);
      a1[4]=tobf(f1.x);  a1[5]=tobf(f1.y);  a1[6]=tobf(f1.z);  a1[7]=tobf(f1.w);
      a1[8]=tobf(f2.x);  a1[9]=tobf(f2.y);  a1[10]=tobf(f2.z); a1[11]=tobf(f2.w);
      a1[12]=tobf(f3.x); a1[13]=tobf(f3.y); a1[14]=tobf(f3.z); a1[15]=tobf(f3.w);
      #pragma unroll
      for (int p8 = 0; p8 < 8; ++p8) {
        const char* fb = s_dyn + ((kt * 8 + p8) << 10);
        BFrag b;
        b.u[0] = *(const uint4*)(fb + co0);
        b.u[1] = *(const uint4*)(fb + co1);
        acc0[p8] = __builtin_amdgcn_wmma_f32_16x16x32_bf16(
            false, a0, false, b.v, (short)0, acc0[p8], false, false);
        acc1[p8] = __builtin_amdgcn_wmma_f32_16x16x32_bf16(
            false, a1, false, b.v, (short)0, acc1[p8], false, false);
      }
    }

    // ---- fold this phase's P-tiles into score partials ---------------------
    #pragma unroll
    for (int p8 = 0; p8 < 8; ++p8) {
      int p = (ph * 8 + p8) * 16 + lm;
      float qb0 = s_q[0][p], qb1 = s_q[1][p];
      float wv = w_att[p];
      #pragma unroll
      for (int i = 0; i < 8; ++i) {
        sacc0[i] += tanhf(acc0[p8][i] + qb0) * wv;
        sacc1[i] += tanhf(acc1[p8][i] + qb1) * wv;
      }
    }
  }

  // ---- reduce scores across the 16 p-lanes, write to LDS -------------------
  #pragma unroll
  for (int off = 1; off < 16; off <<= 1) {
    #pragma unroll
    for (int i = 0; i < 8; ++i) {
      sacc0[i] += __shfl_xor(sacc0[i], off, 32);
      sacc1[i] += __shfl_xor(sacc1[i], off, 32);
    }
  }
  if (lm == 0) {
    #pragma unroll
    for (int i = 0; i < 8; ++i) {
      s_scores[0][wave * 16 + 8 * h + i] = sacc0[i];
      s_scores[1][wave * 16 + 8 * h + i] = sacc1[i];
    }
  }
  __syncthreads();

  // ---- softmax: wave 0 -> n0, wave 1 -> n1 ---------------------------------
  if (wave < 2) {
    float* sc = s_scores[wave];
    float* at = s_att[wave];
    float mx = -1e30f;
    for (int j = lane; j < M_; j += 32) mx = fmaxf(mx, sc[j]);
    #pragma unroll
    for (int off = 16; off >= 1; off >>= 1) mx = fmaxf(mx, __shfl_xor(mx, off, 32));
    float sum = 0.f;
    for (int j = lane; j < M_; j += 32) {
      float e = __expf(sc[j] - mx);
      at[j] = e; sum += e;
    }
    #pragma unroll
    for (int off = 16; off >= 1; off >>= 1) sum += __shfl_xor(sum, off, 32);
    float inv = 1.f / sum;
    for (int j = lane; j < M_; j += 32) at[j] *= inv;
  }
  __syncthreads();

  // ---- context: thread tid -> column d = tid, both rows (tiles L2-hot) -----
  #pragma unroll
  for (int r = 0; r < 2; ++r) {
    const float* memn = memory + (size_t)(n0 + r) * M_ * DM_;
    float c = 0.f;
    for (int m = 0; m < M_; ++m) c += s_att[r][m] * memn[(size_t)m * DM_ + tid];
    concat_u16[(size_t)(n0 + r) * (DM_ + DQ_) + tid] = bfbits(c);
  }
}

// ---------------------------------------------------------------------------
// out = relu(concat @ W_cat + b), bf16 WMMA, one wave per 16x16 tile.
// ---------------------------------------------------------------------------
__global__ void final_gemm_kernel(const __hip_bfloat16* __restrict__ concat_bf,
                                  const char* __restrict__ Wcatpack,
                                  const float* __restrict__ bias,
                                  float* __restrict__ out) {
  int gw   = (blockIdx.x * blockDim.x + threadIdx.x) >> 5;
  int lane = threadIdx.x & 31;
  int lm = lane & 15, h = lane >> 4;
  int nt = gw >> 5;          // 0..127
  int dt = gw & 31;          // 0..31
  int co0 = frag_chunk_off(lane, 0), co1 = frag_chunk_off(lane, 1);
  const char* Arow = (const char*)(concat_bf + (size_t)(nt * 16 + lm) * (DM_ + DQ_));
  v8f acc = {};
  #pragma unroll
  for (int kt = 0; kt < (DM_ + DQ_) / 32; ++kt) {
    BFrag a;
    a.u[0] = *(const uint4*)(Arow + (kt * 32 + 8 * h) * 2);
    a.u[1] = *(const uint4*)(Arow + (kt * 32 + 16 + 8 * h) * 2);
    BFrag b;
    const char* fb = Wcatpack + (size_t)(kt * (DQ_ / 16) + dt) * 1024;
    b.u[0] = *(const uint4*)(fb + co0);
    b.u[1] = *(const uint4*)(fb + co1);
    acc = __builtin_amdgcn_wmma_f32_16x16x32_bf16(false, a.v, false, b.v,
                                                  (short)0, acc, false, false);
  }
  int col = dt * 16 + lm;
  float bs = bias[col];
  #pragma unroll
  for (int i = 0; i < 8; ++i) {
    int row = nt * 16 + i + 8 * h;
    float v = acc[i] + bs;
    out[(size_t)row * DQ_ + col] = v > 0.f ? v : 0.f;
  }
}

// ---------------------------------------------------------------------------
extern "C" void kernel_launch(void* const* d_in, const int* in_sizes, int n_in,
                              void* d_out, int out_size, void* d_ws, size_t ws_size,
                              hipStream_t stream) {
  const float* query   = (const float*)d_in[0];   // [N, DQ]
  const float* memory  = (const float*)d_in[1];   // [N, M, DM]
  const float* W_att_w = (const float*)d_in[2];   // [DQ, P]
  const float* W_att_b = (const float*)d_in[3];   // [P]
  const float* U_att_w = (const float*)d_in[4];   // [DM, P]
  const float* U_att_b = (const float*)d_in[5];   // [P]
  const float* w_att_w = (const float*)d_in[6];   // [P]
  // d_in[7] = w_att_b : constant shift, cancels inside softmax
  const float* W_cat_w = (const float*)d_in[8];   // [DQ+DM, DQ]
  const float* W_cat_b = (const float*)d_in[9];   // [DQ]
  float* out = (float*)d_out;

  char* ws = (char*)d_ws;
  float*          qws      = (float*)(ws);                                  // 2 MB
  unsigned*       Upack    = (unsigned*)(ws + (2u << 20));                  // 256 KB
  unsigned*       Wattpack = (unsigned*)(ws + (2u << 20) + (256u << 10));   // 256 KB
  unsigned*       Wcatpack = (unsigned*)(ws + (2u << 20) + (512u << 10));   // 1 MB
  unsigned short* concat   = (unsigned short*)(ws + (3u << 20) + (512u << 10)); // 4 MB

  // One-off packing / conversion (all tiny, L2-resident afterwards)
  pack_b_kernel<<<256, 256, 0, stream>>>(U_att_w, Upack, DM_, P_);
  pack_b_kernel<<<256, 256, 0, stream>>>(W_att_w, Wattpack, DQ_, P_);
  pack_b_kernel<<<1024, 256, 0, stream>>>(W_cat_w, Wcatpack, DM_ + DQ_, DQ_);
  cvt_query_kernel<<<2048, 256, 0, stream>>>(query, (unsigned*)concat);

  // q = query @ W_att + (W_att_b + U_att_b)
  q_gemm_kernel<<<256, 256, 0, stream>>>((const __hip_bfloat16*)concat,
                                         (const char*)Wattpack,
                                         W_att_b, U_att_b, qws);

  // Fused: m-GEMM + tanh scores + softmax + context; 2 rows per WG,
  // U_att staged to LDS via async copies (128KB dynamic LDS per phase).
  fused_kernel<<<1024, 512, 131072, stream>>>(memory, qws, (const char*)Upack,
                                              w_att_w, concat);

  // out = relu([context | query] @ W_cat + b)
  final_gemm_kernel<<<512, 256, 0, stream>>>((const __hip_bfloat16*)concat,
                                             (const char*)Wcatpack,
                                             W_cat_b, out);
}
